// MarkoNetv1_26079041421743
// MI455X (gfx1250) — matrix operand order repaired and checked by reading, not compiled, not run on hardware
//
#include <hip/hip_runtime.h>
#include <cstddef>
#include <cstdint>

// ---------------------------------------------------------------------------
// MarkoNet forward on MI455X (gfx1250, wave32).
// All matmuls run on v_wmma_f32_16x16x32_bf16 with f32 accumulation.
// Activations + weights live in bf16 workspace buffers (identical numerics to
// converting at the WMMA inputs, but half the bytes and no per-tile cvt work);
// weights are converted from the fp32 inputs once per launch.
// Attention is fused flash-style (no materialized score matrix).
// All buffers padded to 32-row multiples and pre-zeroed -> every hot-loop
// load/store is unconditional and vectorized (b128).
// ---------------------------------------------------------------------------

#define HS      128
#define NHEADS  4
#define HDIM    32
#define NN      2304      // 48*48 graph tokens
#define PP      128       // pauli rows
#define OO      4096
#define ITERS   5
#define NTILE   2         // 16-col C tiles per wave in the GEMM

typedef __attribute__((ext_vector_type(16))) __bf16 v16bf;
typedef __attribute__((ext_vector_type(8)))  __bf16 v8bf;
typedef __attribute__((ext_vector_type(8)))  float  v8f;

static __device__ __forceinline__ v16bf cat8(v8bf lo, v8bf hi) {
  v16bf r;
#pragma unroll
  for (int e = 0; e < 8; ++e) { r[e] = lo[e]; r[8 + e] = hi[e]; }
  return r;
}

// ---- WMMA fragment loaders (contiguous k order: k = 16*half + element, wave32) ----------
// All loads unconditional: callers guarantee padded, readable bf16 buffers.

// 16-bit A fragment 16x32 (MxK), A row-major bf16, ld = HS.
// Lane half 0 holds K [k0+0..15]; half 1 holds [k0+16..31].
__device__ __forceinline__ v16bf load_a_frag(const __bf16* __restrict__ A,
                                             int m0, int k0, int lane) {
  const int m  = m0 + (lane & 15);
  const int kb = k0 + ((lane >> 4) << 4);
  const __bf16* p = A + (size_t)m * HS + kb;
  return cat8(*(const v8bf*)p, *(const v8bf*)(p + 8));    // 2x b128
}

// 16-bit B fragment 32x16 (KxN) for B^T layout: Bt is [N,K] row-major, ld=HS.
// Lane holds column n = n0+(lane&15); half 0 -> K k0..k0+15, half 1 -> +16..31.
__device__ __forceinline__ v16bf load_bt_frag(const __bf16* __restrict__ Bt,
                                              int n0, int k0, int lane) {
  const int n = n0 + (lane & 15);
  const __bf16* p = Bt + (size_t)n * HS + k0 + ((lane >> 4) << 4);
  return cat8(*(const v8bf*)p, *(const v8bf*)(p + 8));    // 2x b128
}

// B fragment from row-major [L,HS] bf16 without transpose (used for V):
// element B[k][n] = V[(k0+k)*HS + n]; coalesced 16-lane rows, 16 b16 per lane.
__device__ __forceinline__ v16bf load_v_frag(const __bf16* __restrict__ V,
                                             int k0, int n, int lane) {
  const __bf16* p = V + (size_t)(k0 + ((lane >> 4) << 4)) * HS + n;
  v16bf b;
#pragma unroll
  for (int e = 0; e < 16; ++e) b[e] = p[(size_t)e * HS];
  return b;
}

// ---- Specialized projection GEMM -------------------------------------------
// C[M,128] = A[M,128] @ W[128,128]^T + bias (bias fp32, added pre-truncation).
// One wave per 16x(16*NTILE) tile, 4 m-tiles per block.
__global__ void mn_gemm_kernel(const __bf16* __restrict__ A,
                               const __bf16* __restrict__ W,
                               const float* __restrict__ bias,
                               __bf16* __restrict__ C, int Mtiles) {
  const int wave = threadIdx.x >> 5;
  const int lane = threadIdx.x & 31;
  const int mtile = blockIdx.x * 4 + wave;
  if (mtile >= Mtiles) return;
  const int m0 = mtile * 16;
  const int n0 = blockIdx.y * (16 * NTILE);

  v8f acc[NTILE];
#pragma unroll
  for (int t = 0; t < NTILE; ++t) acc[t] = v8f{};

#pragma unroll
  for (int k0 = 0; k0 < HS; k0 += 32) {
    v16bf a = load_a_frag(A, m0, k0, lane);
#pragma unroll
    for (int t = 0; t < NTILE; ++t) {
      v16bf b = load_bt_frag(W, n0 + t * 16, k0, lane);
      acc[t] = __builtin_amdgcn_wmma_f32_16x16x32_bf16(false, a, false, b,
                                                       (short)0, acc[t],
                                                       false, false);
    }
  }

  const int rbase = (lane >> 4) * 8;   // C frag: VGPR r -> row r (+8 upper half)
#pragma unroll
  for (int t = 0; t < NTILE; ++t) {
    const int n = n0 + t * 16 + (lane & 15);
    const float bv = bias[n];
#pragma unroll
    for (int r = 0; r < 8; ++r)
      C[(size_t)(m0 + rbase + r) * HS + n] = (__bf16)(acc[t][r] + bv);
  }
}

// ---- Fused flash-style MHA core --------------------------------------------
// Q,K,V,O: [*,128] bf16, heads concatenated, padded to 32-row multiples with
// zeros.  blockDim=128, wave w == head w, one 16-query tile per block.
__global__ void mn_flash_mha_kernel(const __bf16* __restrict__ Q,
                                    const __bf16* __restrict__ K,
                                    const __bf16* __restrict__ V,
                                    __bf16* __restrict__ O, int Lk) {
  __shared__ __bf16 psh[NHEADS][16 * 32];
  const int tid  = threadIdx.x;
  const int wave = tid >> 5;           // head index
  const int lane = tid & 31;
  const int coff = wave * HDIM;        // column offset of this head
  const int m0   = blockIdx.x * 16;
  const float scale = 0.17677669529663687f;  // 1/sqrt(32)
  const int ncol = lane & 15;
  const int half = lane >> 4;

  // Q tile A-fragment (16 queries x 32 head-dim), reused across all key tiles.
  const v16bf qa = load_a_frag(Q + coff, m0, 0, lane);

  float mrun[8], lrun[8];
  v8f o0 = {}, o1 = {};
#pragma unroll
  for (int r = 0; r < 8; ++r) { mrun[r] = -3.0e38f; lrun[r] = 0.0f; }

  for (int n0 = 0; n0 < Lk; n0 += 32) {
    // S = Q_h @ K_h^T for 32 keys: two 16x16 WMMAs (K^T == B^T of row-major K).
    v16bf kb0 = load_bt_frag(K + coff, n0, 0, lane);
    v16bf kb1 = load_bt_frag(K + coff, n0 + 16, 0, lane);
    v8f z = {};
    v8f s0 = __builtin_amdgcn_wmma_f32_16x16x32_bf16(false, qa, false, kb0,
                                                     (short)0, z, false, false);
    v8f s1 = __builtin_amdgcn_wmma_f32_16x16x32_bf16(false, qa, false, kb1,
                                                     (short)0, z, false, false);
    // Online softmax across these 32 columns (key mask on column index only).
#pragma unroll
    for (int r = 0; r < 8; ++r) {
      float x0 = (n0 + ncol      < Lk) ? s0[r] * scale : -3.0e38f;
      float x1 = (n0 + 16 + ncol < Lk) ? s1[r] * scale : -3.0e38f;
      float mx = fmaxf(x0, x1);
#pragma unroll
      for (int d = 1; d < 16; d <<= 1)       // row-reduce within 16-lane half
        mx = fmaxf(mx, __shfl_xor(mx, d, 32));
      const float mnew = fmaxf(mrun[r], mx);
      const float p0 = __expf(x0 - mnew);
      const float p1 = __expf(x1 - mnew);
      float ps = p0 + p1;
#pragma unroll
      for (int d = 1; d < 16; d <<= 1)
        ps += __shfl_xor(ps, d, 32);
      const float alpha = __expf(mrun[r] - mnew);
      lrun[r] = lrun[r] * alpha + ps;
      mrun[r] = mnew;
      o0[r] *= alpha;
      o1[r] *= alpha;
      const int prow = r + 8 * half;         // C-frag row id
      psh[wave][prow * 32 + ncol]      = (__bf16)p0;
      psh[wave][prow * 32 + 16 + ncol] = (__bf16)p1;
    }
    __syncthreads();
    // P (16x32 bf16) -> A fragment via two b128 LDS loads per lane.
    v16bf pa;
    {
      const int pm = lane & 15;
      const int kb = (lane >> 4) << 4;
      pa = cat8(*(const v8bf*)&psh[wave][pm * 32 + kb],
                *(const v8bf*)&psh[wave][pm * 32 + kb + 8]);
    }
    v16bf vb0 = load_v_frag(V, n0, coff + (lane & 15), lane);
    v16bf vb1 = load_v_frag(V, n0, coff + 16 + (lane & 15), lane);
    o0 = __builtin_amdgcn_wmma_f32_16x16x32_bf16(false, pa, false, vb0,
                                                 (short)0, o0, false, false);
    o1 = __builtin_amdgcn_wmma_f32_16x16x32_bf16(false, pa, false, vb1,
                                                 (short)0, o1, false, false);
    __syncthreads();
  }

  // Normalize and store (O padded -> unguarded).
#pragma unroll
  for (int r = 0; r < 8; ++r) {
    const float inv = (lrun[r] > 0.0f) ? (1.0f / lrun[r]) : 0.0f;
    const int m = m0 + r + 8 * half;
    O[(size_t)m * HS + coff + ncol]      = (__bf16)(o0[r] * inv);
    O[(size_t)m * HS + coff + 16 + ncol] = (__bf16)(o1[r] * inv);
  }
}

// ---- Small helper kernels ---------------------------------------------------
// embedding gather-sum: out[p,h] = sum_{o: mask} emb[o,h]  (f32 accum -> bf16)
__global__ void mn_embed_kernel(const int* __restrict__ p_ops,
                                const float* __restrict__ emb,
                                __bf16* __restrict__ out) {  // out: [PP,HS]
  const int p = blockIdx.x;
  const int h = threadIdx.x;
  const int* row = p_ops + (size_t)p * OO;
  float s = 0.0f;
  for (int o = 0; o < OO; o += 4) {
    const int4 mv = *(const int4*)(row + o);
    s += (mv.x != 0) ? emb[(size_t)(o + 0) * HS + h] : 0.0f;
    s += (mv.y != 0) ? emb[(size_t)(o + 1) * HS + h] : 0.0f;
    s += (mv.z != 0) ? emb[(size_t)(o + 2) * HS + h] : 0.0f;
    s += (mv.w != 0) ? emb[(size_t)(o + 3) * HS + h] : 0.0f;
  }
  out[(size_t)p * HS + h] = (__bf16)s;
}

__global__ void mn_cvtbf_kernel(__bf16* __restrict__ dst,
                                const float* __restrict__ src, int n) {
  const int i = blockIdx.x * 256 + threadIdx.x;
  if (i < n) dst[i] = (__bf16)src[i];
}

__global__ void mn_copyu4_kernel(uint4* __restrict__ dst,
                                 const uint4* __restrict__ src, int n4) {
  const int i = blockIdx.x * 256 + threadIdx.x;
  if (i < n4) dst[i] = src[i];
}

// broadcast one 128-col bf16 row into `rows` rows (dword-wide).
__global__ void mn_bcast_row_kernel(unsigned int* __restrict__ dst,
                                    const unsigned int* __restrict__ src,
                                    int rows) {
  const int i = blockIdx.x * 256 + threadIdx.x;
  if (i < rows * (HS / 2)) dst[i] = src[i & (HS / 2 - 1)];
}

__global__ void mn_zero4_kernel(float4* __restrict__ p, int n4) {
  const int i = blockIdx.x * 256 + threadIdx.x;
  if (i < n4) p[i] = float4{0.0f, 0.0f, 0.0f, 0.0f};
}

// out[m] = dot(X[m,:], fW) + fb ; one wave per row, 4 cols per lane.
__global__ void mn_final_kernel(const __bf16* __restrict__ X,
                                const float* __restrict__ fW,
                                const float* __restrict__ fb,
                                float* __restrict__ out) {
  const int row  = blockIdx.x * 4 + (threadIdx.x >> 5);
  const int lane = threadIdx.x & 31;
  const __bf16* xr = X + (size_t)row * HS + lane * 4;
  const float4 w = ((const float4*)fW)[lane];
  float s = (float)xr[0] * w.x + (float)xr[1] * w.y +
            (float)xr[2] * w.z + (float)xr[3] * w.w;
#pragma unroll
  for (int d = 16; d >= 1; d >>= 1) s += __shfl_xor(s, d, 32);
  if (lane == 0) out[row] = s + fb[0];
}

// ---------------------------------------------------------------------------
extern "C" void kernel_launch(void* const* d_in, const int* in_sizes, int n_in,
                              void* d_out, int out_size, void* d_ws, size_t ws_size,
                              hipStream_t stream) {
  (void)in_sizes; (void)n_in; (void)out_size; (void)ws_size;

  const int*   p_ops      = (const int*)  d_in[0];
  const float* emb        = (const float*)d_in[1];
  const float* init_instr = (const float*)d_in[2];
  const float* init_graph = (const float*)d_in[3];
  const float* p_Wi  = (const float*)d_in[4];
  const float* p_bi  = (const float*)d_in[5];
  const float* p_Wo  = (const float*)d_in[6];
  const float* p_bo  = (const float*)d_in[7];
  const float* pg_Wi = (const float*)d_in[8];
  const float* pg_bi = (const float*)d_in[9];
  const float* pg_Wo = (const float*)d_in[10];
  const float* pg_bo = (const float*)d_in[11];
  const float* g_Wi  = (const float*)d_in[12];
  const float* g_bi  = (const float*)d_in[13];
  const float* g_Wo  = (const float*)d_in[14];
  const float* g_bo  = (const float*)d_in[15];
  const float* g_bk  = (const float*)d_in[16];
  const float* g_bv  = (const float*)d_in[17];
  const float* f_W   = (const float*)d_in[18];
  const float* f_b   = (const float*)d_in[19];

  // Workspace partition: bf16 activations (32-row padded) + bf16 weights.
  const int PE_R = 160;    // p_embs   (129 used)
  const int NN_R = NN;     // curr, graphe (2304, 32-aligned already)
  const int NG_R = 2336;   // ngbuf/ngout/AO (2305 used, <=2320 written)
  const int QB_R = 2336;   // Q buffer
  const int KB_R = 2336;   // K/V buffers (incl. appended bias row)
  const int ONE_R = 16;    // single-row results

  __bf16* w = (__bf16*)d_ws;
  __bf16* p_embs = w; w += PE_R * HS;
  __bf16* curr   = w; w += NN_R * HS;
  __bf16* graphe = w; w += NN_R * HS;
  __bf16* ngbuf  = w; w += NG_R * HS;
  __bf16* ngout  = w; w += NG_R * HS;
  __bf16* Qb     = w; w += QB_R * HS;
  __bf16* Kb     = w; w += KB_R * HS;
  __bf16* Vb     = w; w += KB_R * HS;
  __bf16* AO     = w; w += NG_R * HS;
  __bf16* pA0    = w; w += ONE_R * HS;
  __bf16* instr  = w; w += ONE_R * HS;
  // bf16 weight copies (converted once per launch)
  __bf16* wp_Wi  = w; w += 3 * HS * HS;
  __bf16* wp_Wo  = w; w += HS * HS;
  __bf16* wpg_Wi = w; w += 3 * HS * HS;
  __bf16* wpg_Wo = w; w += HS * HS;
  __bf16* wg_Wi  = w; w += 3 * HS * HS;
  __bf16* wg_Wo  = w; w += HS * HS;
  const size_t total_bytes = (size_t)((char*)w - (char*)d_ws);

  // Zero all workspace (pads must read as 0; keeps every call identical).
  {
    const int n4 = (int)(total_bytes / 16);
    mn_zero4_kernel<<<(n4 + 255) / 256, 256, 0, stream>>>((float4*)d_ws, n4);
  }

  auto cvt = [&](__bf16* dst, const float* src, int n) {
    mn_cvtbf_kernel<<<(n + 255) / 256, 256, 0, stream>>>(dst, src, n);
  };
  auto copyh = [&](__bf16* dst, const __bf16* src, int n) {  // n % 8 == 0
    const int n4 = n / 8;
    mn_copyu4_kernel<<<(n4 + 255) / 256, 256, 0, stream>>>(
        (uint4*)dst, (const uint4*)src, n4);
  };
  auto gemm = [&](const __bf16* A, const __bf16* W, const float* bias,
                  __bf16* C, int M) {
    const int mtiles = (M + 15) / 16;
    dim3 grid((mtiles + 3) / 4, HS / (16 * NTILE));
    mn_gemm_kernel<<<grid, 128, 0, stream>>>(A, W, bias, C, mtiles);
  };
  auto mha = [&](const __bf16* xq, int Lq, const __bf16* xkv, int Lkv,
                 const __bf16* Wi, const float* bi, const __bf16* Wo,
                 const float* bo, const float* bk, const float* bv,
                 __bf16* out) {
    gemm(xq,  Wi,               bi,          Qb, Lq);
    gemm(xkv, Wi + HS * HS,     bi + HS,     Kb, Lkv);
    gemm(xkv, Wi + 2 * HS * HS, bi + 2 * HS, Vb, Lkv);
    int Lk = Lkv;
    if (bk != nullptr) {                 // add_bias_kv: one extra kv token
      cvt(Kb + (size_t)Lkv * HS, bk, HS);
      cvt(Vb + (size_t)Lkv * HS, bv, HS);
      Lk = Lkv + 1;
    }
    mn_flash_mha_kernel<<<(Lq + 15) / 16, 32 * NHEADS, 0, stream>>>(
        Qb, Kb, Vb, AO, Lk);
    gemm(AO, Wo, bo, out, Lq);
  };

  // ---- weights -> bf16 (once per launch) ----
  cvt(wp_Wi,  p_Wi,  3 * HS * HS);  cvt(wp_Wo,  p_Wo,  HS * HS);
  cvt(wpg_Wi, pg_Wi, 3 * HS * HS);  cvt(wpg_Wo, pg_Wo, HS * HS);
  cvt(wg_Wi,  g_Wi,  3 * HS * HS);  cvt(wg_Wo,  g_Wo,  HS * HS);

  // ---- init state (fully re-derived from inputs every call) ----
  cvt(p_embs, init_instr, HS);                                    // row 0
  mn_embed_kernel<<<PP, HS, 0, stream>>>(p_ops, emb, p_embs + HS);
  cvt(curr, init_graph, NN * HS);

  for (int it = 0; it < ITERS; ++it) {
    // 1) p self-attention; only row 0 of the result is ever consumed.
    mha(p_embs, 1, p_embs, PP + 1, wp_Wi, p_bi, wp_Wo, p_bo,
        nullptr, nullptr, pA0);
    // 2) graph self-attention (with bias kv token).
    mha(curr, NN, curr, NN, wg_Wi, g_bi, wg_Wo, g_bo, g_bk, g_bv, graphe);
    // 3) instr cross-attention: query = p_attn[0], kv = graph_emb.
    mha(pA0, 1, graphe, NN, wpg_Wi, pg_bi, wpg_Wo, pg_bo,
        nullptr, nullptr, instr);
    // 4) ng = [instr; graph_emb]; self-attention; curr = result[1:].
    copyh(ngbuf, instr, HS);
    copyh(ngbuf + HS, graphe, NN * HS);
    mha(ngbuf, NN + 1, ngbuf, NN + 1, wg_Wi, g_bi, wg_Wo, g_bo,
        g_bk, g_bv, ngout);
    copyh(curr, ngout + HS, NN * HS);
    // 5) p_embs <- broadcast(instr)  (torch row-assignment broadcast).
    {
      const int nu = (PP + 1) * (HS / 2);
      mn_bcast_row_kernel<<<(nu + 255) / 256, 256, 0, stream>>>(
          (unsigned int*)p_embs, (const unsigned int*)instr, PP + 1);
    }
  }

  // final head: out[m] = curr[m,:] . f_W + f_b  -> [2304]
  mn_final_kernel<<<NN / 4, 128, 0, stream>>>(curr, f_W, f_b, (float*)d_out);
}
